// MultiHeadAttention_45234595561826
// MI455X (gfx1250) — matrix operand order repaired and checked by reading, not compiled, not run on hardware
//
#include <hip/hip_runtime.h>

// ---------- CDNA5 WMMA types ----------
typedef __attribute__((ext_vector_type(16))) __bf16 v16bf;
typedef __attribute__((ext_vector_type(8)))  float  v8f;

union BFrag {
    v16bf v;
    unsigned short u[16];
};

__device__ __forceinline__ unsigned short f2bf(float f) {
    union { float f; unsigned u; } x; x.f = f;
    unsigned r = x.u + 0x7FFFu + ((x.u >> 16) & 1u);   // round-to-nearest-even
    return (unsigned short)(r >> 16);
}
__device__ __forceinline__ float bf2f(unsigned short h) {
    union { unsigned u; float f; } x; x.u = ((unsigned)h) << 16;
    return x.f;
}

// ---------- gfx1250 async global->LDS path (guarded; falls back to uint4 copies) ----
#ifndef __has_builtin
#define __has_builtin(x) 0
#endif

#if __has_builtin(__builtin_amdgcn_global_load_async_to_lds_b128)
#define ASYNC_LDS 1
#else
#define ASYNC_LDS 0
#endif

#if ASYNC_LDS
// builtin signature (from hipcc diagnostic): params are pointers to
// 'int __attribute__((vector_size(16)))' in addrspace(1) / addrspace(3)
typedef int v4i_g __attribute__((vector_size(16)));
typedef __attribute__((address_space(1))) v4i_g* gas1_v4;
typedef __attribute__((address_space(3))) v4i_g* gas3_v4;

__device__ __forceinline__ void async_cp16(const void* g, void* l) {
    // 16B per lane, Memory -> LDS, tracked by ASYNCcnt
    __builtin_amdgcn_global_load_async_to_lds_b128(
        (gas1_v4)(size_t)g, (gas3_v4)(unsigned)(size_t)l, 0, 0);
}
__device__ __forceinline__ void wait_async0() {
#if __has_builtin(__builtin_amdgcn_s_wait_asynccnt)
    __builtin_amdgcn_s_wait_asynccnt(0);
#else
    asm volatile("s_wait_asynccnt 0" ::: "memory");
#endif
}
#define WAIT_ASYNC() wait_async0()
#else
#define WAIT_ASYNC() ((void)0)
#endif

// =====================================================================
// fp32 -> bf16 bulk convert (one shot; removes conversions from GEMM hot loop)
// =====================================================================
__global__ void __launch_bounds__(256)
f32_to_bf16_kernel(const float* __restrict__ src, unsigned short* __restrict__ dst, int n4)
{
    int i = blockIdx.x * blockDim.x + threadIdx.x;
    if (i >= n4) return;
    float4 d = ((const float4*)src)[i];
    uint2 r;
    r.x = (unsigned)f2bf(d.x) | ((unsigned)f2bf(d.y) << 16);
    r.y = (unsigned)f2bf(d.z) | ((unsigned)f2bf(d.w) << 16);
    ((uint2*)dst)[i] = r;
}

// =====================================================================
// GEMM: out[m,n] = sum_k A[m,k] * W[n,k] + bias[n]    (A, W already bf16)
// Block tile 128x128, K-step 32, 256 threads = 8 waves (2x4), wave tile 64x32.
// Double-buffered LDS fed by async global->LDS (ASYNCcnt) when available.
// =====================================================================
__device__ __forceinline__ void stage_tile_128x32(const unsigned short* __restrict__ g,
                                                  int row0, int K, int k0,
                                                  unsigned short (*lds)[40], int tid)
{
#pragma unroll
    for (int it = 0; it < 2; ++it) {
        int e = (tid + it * 256) * 8;     // 8 ushorts = 16B per lane
        int r = e >> 5, cc = e & 31;
        const unsigned short* src = g + (size_t)(row0 + r) * K + (k0 + cc);
#if ASYNC_LDS
        async_cp16(src, &lds[r][cc]);
#else
        *(uint4*)&lds[r][cc] = *(const uint4*)src;
#endif
    }
}

template <bool OUT_F32>
__global__ void __launch_bounds__(256)
gemm_bt_bf16_kernel(const unsigned short* __restrict__ A,
                    const unsigned short* __restrict__ W,
                    const float* __restrict__ bias,
                    float* __restrict__ outF,
                    unsigned short* __restrict__ outB,
                    int M, int N, int K)
{
    // row stride 40 ushorts = 80B (16B multiple -> aligned b128 async stores)
    __shared__ __attribute__((aligned(16))) unsigned short As[2][128][40];
    __shared__ __attribute__((aligned(16))) unsigned short Bs[2][128][40];

    const int tid  = threadIdx.x;
    const int lane = tid & 31;
    const int w    = tid >> 5;      // wave 0..7
    const int wm   = w >> 2;        // 0..1
    const int wn   = w & 3;         // 0..3
    const int hh   = lane >> 4;     // lane half
    const int l16  = lane & 15;
    const int m0   = blockIdx.x * 128;
    const int n0   = blockIdx.y * 128;

    v8f c[4][2] = {};

    // prologue: stage k0 = 0 into buffer 0
    stage_tile_128x32(A, m0, K, 0, As[0], tid);
    stage_tile_128x32(W, n0, K, 0, Bs[0], tid);
    WAIT_ASYNC();
    __syncthreads();

    int buf = 0;
    for (int k0 = 0; k0 < K; k0 += 32) {
        // async prefetch of next K tile into the other buffer
        if (k0 + 32 < K) {
            stage_tile_128x32(A, m0, K, k0 + 32, As[buf ^ 1], tid);
            stage_tile_128x32(W, n0, K, k0 + 32, Bs[buf ^ 1], tid);
        }

        // ---- build fragments from resident buffer, run 4x2 WMMAs ----
        BFrag a[4], b[2];
#pragma unroll
        for (int mi = 0; mi < 4; ++mi) {
            int m = wm * 64 + mi * 16 + l16;       // A: lane carries row M
#pragma unroll
            for (int e = 0; e < 16; ++e) {
                int kk = e + hh * 16;                         // contiguous K: 16 per lane half
                a[mi].u[e] = As[buf][m][kk];
            }
        }
#pragma unroll
        for (int ni = 0; ni < 2; ++ni) {
            int n = wn * 32 + ni * 16 + l16;       // B: lane carries col N
#pragma unroll
            for (int e = 0; e < 16; ++e) {
                int kk = e + hh * 16;              // B 32x16: K split 16/16 across halves
                b[ni].u[e] = Bs[buf][n][kk];
            }
        }
#pragma unroll
        for (int mi = 0; mi < 4; ++mi)
#pragma unroll
            for (int ni = 0; ni < 2; ++ni)
                c[mi][ni] = __builtin_amdgcn_wmma_f32_16x16x32_bf16(
                    false, a[mi].v, false, b[ni].v, (short)0, c[mi][ni], false, false);

        WAIT_ASYNC();          // next tile resident
        __syncthreads();       // everyone done reading current buffer
        buf ^= 1;
    }

    // ---- epilogue: bias add + store ----
#pragma unroll
    for (int ni = 0; ni < 2; ++ni) {
        int n = n0 + wn * 32 + ni * 16 + l16;
        float bn = bias[n];
#pragma unroll
        for (int mi = 0; mi < 4; ++mi) {
#pragma unroll
            for (int v = 0; v < 8; ++v) {
                int m = m0 + wm * 64 + mi * 16 + hh * 8 + v;   // C/D: rows in VGPRs
                float val = c[mi][ni][v] + bn;
                if (OUT_F32) outF[(size_t)m * N + n] = val;
                else         outB[(size_t)m * N + n] = f2bf(val);
            }
        }
    }
}

// =====================================================================
// RoPE on bf16 Q and K in-place. Layout [S, H*HD]; pairs (i, i+32).
// =====================================================================
__global__ void __launch_bounds__(256)
rope_kernel(unsigned short* __restrict__ q, unsigned short* __restrict__ k,
            int S, int Hn, int HD)
{
    int idx = blockIdx.x * blockDim.x + threadIdx.x;
    const int half  = HD >> 1;                // 32
    const int total = S * Hn * half;          // per tensor
    unsigned short* p = q;
    if (idx >= total) { p = k; idx -= total; }
    if (idx >= total) return;

    int i    = idx % half;
    int rest = idx / half;
    int h    = rest % Hn;
    int s    = rest / Hn;

    float inv = __powf(10000.0f, -(float)i / (float)half);
    float ang = (float)s * inv;
    float sn, cs;
    __sincosf(ang, &sn, &cs);

    size_t base = ((size_t)s * Hn + h) * HD + i;
    float x1 = bf2f(p[base]);
    float x2 = bf2f(p[base + half]);
    p[base]        = f2bf(x1 * cs - x2 * sn);
    p[base + half] = f2bf(x2 * cs + x1 * sn);
}

// =====================================================================
// Flash attention, causal. Grid (S/64, H), 128 threads = 4 waves.
// K tiles: async global->LDS DMA (row-major, matches B-fragment reads).
// V tiles: staged TRANSPOSED (VtT[hd][key]) so P*V B-fragments are also
// contiguous ds_load_b128 instead of per-element column gathers.
// QK^T and P*V on v_wmma_f32_16x16x32_bf16; online softmax via wave32
// shfl_xor butterflies (rows in VGPR index, cols in lanes).
// =====================================================================
__device__ __forceinline__ void stage_kv_64x64(const unsigned short* __restrict__ Kg,
                                               const unsigned short* __restrict__ Vg,
                                               size_t gbase, int D,
                                               unsigned short (*Kt)[72],
                                               unsigned short (*VtT)[72], int tid)
{
#pragma unroll
    for (int it = 0; it < 4; ++it) {
        int e = (tid + it * 128) * 8;     // 8 ushorts = 16B per lane
        int r = e >> 6, cc = e & 63;      // r = key row, cc = hd base
        size_t g = gbase + (size_t)r * D + cc;
#if ASYNC_LDS
        async_cp16(Kg + g, &Kt[r][cc]);
#else
        *(uint4*)&Kt[r][cc] = *(const uint4*)(Kg + g);
#endif
        // V: load 8 hd-contiguous values of key r, scatter transposed into LDS
        uint4 vv = *(const uint4*)(Vg + g);
        const unsigned short* vs = (const unsigned short*)&vv;
#pragma unroll
        for (int j = 0; j < 8; ++j)
            VtT[cc + j][r] = vs[j];
    }
}

__global__ void __launch_bounds__(128)
flash_attn_kernel(const unsigned short* __restrict__ Q,
                  const unsigned short* __restrict__ Kg,
                  const unsigned short* __restrict__ Vg,
                  unsigned short* __restrict__ Og,
                  int S, int D)
{
    // row stride 72 ushorts = 144B (16B multiple)
    __shared__ __attribute__((aligned(16))) unsigned short Kt[2][64][72];   // [key][hd]
    __shared__ __attribute__((aligned(16))) unsigned short VtT[2][64][72];  // [hd][key]
    __shared__ __attribute__((aligned(16))) unsigned short Pt[4][16][72];   // per-wave P

    const int tid  = threadIdx.x;
    const int lane = tid & 31;
    const int w    = tid >> 5;     // wave 0..3
    const int hh   = lane >> 4;
    const int l16  = lane & 15;
    const int h    = blockIdx.y;
    const int qw   = blockIdx.x * 64 + w * 16;  // this wave's first query row

    // Q fragments for this wave's 16 rows (two K-steps covering hd 0..63)
    BFrag aq[2];
    {
        const unsigned short* qrow = Q + (size_t)(qw + l16) * D + h * 64;
#pragma unroll
        for (int st = 0; st < 2; ++st)
#pragma unroll
            for (int e = 0; e < 16; ++e) {
                int kk = st * 32 + e + hh * 16;
                aq[st].u[e] = qrow[kk];
            }
    }

    float rmax[8], rsum[8];
#pragma unroll
    for (int v = 0; v < 8; ++v) { rmax[v] = -3.0e38f; rsum[v] = 0.0f; }
    v8f o[4] = {};

    const int ntiles = blockIdx.x + 1;   // causal: only KV tiles <= query tile

    // prologue: tile 0 into buffer 0
    stage_kv_64x64(Kg, Vg, (size_t)h * 64, D, Kt[0], VtT[0], tid);
    WAIT_ASYNC();
    __syncthreads();

    int buf = 0;
    for (int t = 0; t < ntiles; ++t) {
        // prefetch next KV tile into the other buffer (K async, V transposed)
        if (t + 1 < ntiles)
            stage_kv_64x64(Kg, Vg, (size_t)((t + 1) * 64) * D + h * 64, D,
                           Kt[buf ^ 1], VtT[buf ^ 1], tid);

        // ---- scores S = Q K^T  (4 key sub-tiles x 2 K-steps = 8 WMMA) ----
        v8f sc[4] = {};
#pragma unroll
        for (int nt = 0; nt < 4; ++nt) {
#pragma unroll
            for (int st = 0; st < 2; ++st) {
                BFrag bk;
                int n = nt * 16 + l16;                     // key column
#pragma unroll
                for (int e = 0; e < 16; ++e) {
                    int kk = st * 32 + e + hh * 16;        // hd index
                    bk.u[e] = Kt[buf][n][kk];
                }
                sc[nt] = __builtin_amdgcn_wmma_f32_16x16x32_bf16(
                    false, aq[st].v, false, bk.v, (short)0, sc[nt], false, false);
            }
        }

        // ---- scale + causal mask ----
        const float scale = 0.125f;                        // 1/sqrt(64)
#pragma unroll
        for (int nt = 0; nt < 4; ++nt) {
            int key = t * 64 + nt * 16 + l16;
#pragma unroll
            for (int v = 0; v < 8; ++v) {
                int qq = qw + hh * 8 + v;
                float sv = sc[nt][v] * scale;
                sc[nt][v] = (key <= qq) ? sv : -3.0e38f;
            }
        }

        // ---- online softmax (row reductions across 16 lanes) ----
        float alpha[8];
#pragma unroll
        for (int v = 0; v < 8; ++v) {
            float mx = fmaxf(fmaxf(sc[0][v], sc[1][v]), fmaxf(sc[2][v], sc[3][v]));
#pragma unroll
            for (int msk = 1; msk < 16; msk <<= 1)
                mx = fmaxf(mx, __shfl_xor(mx, msk, 32));
            float mnew = fmaxf(rmax[v], mx);
            alpha[v] = __expf(rmax[v] - mnew);
            rmax[v]  = mnew;
            float lsum = 0.0f;
#pragma unroll
            for (int nt = 0; nt < 4; ++nt) {
                float pv = __expf(sc[nt][v] - mnew);
                sc[nt][v] = pv;
                lsum += pv;
            }
#pragma unroll
            for (int msk = 1; msk < 16; msk <<= 1)
                lsum += __shfl_xor(lsum, msk, 32);
            rsum[v] = rsum[v] * alpha[v] + lsum;
        }

        // ---- rescale running O ----
#pragma unroll
        for (int j = 0; j < 4; ++j)
#pragma unroll
            for (int v = 0; v < 8; ++v)
                o[j][v] *= alpha[v];

        // ---- P (C-layout) -> LDS -> A-fragment transpose ----
#pragma unroll
        for (int nt = 0; nt < 4; ++nt)
#pragma unroll
            for (int v = 0; v < 8; ++v)
                Pt[w][hh * 8 + v][nt * 16 + l16] = f2bf(sc[nt][v]);
        __syncthreads();

        // ---- O += P @ V  (2 K-steps x 4 hd tiles = 8 WMMA) ----
#pragma unroll
        for (int st = 0; st < 2; ++st) {
            BFrag ap;
#pragma unroll
            for (int e = 0; e < 16; ++e) {
                int kk = st * 32 + e + hh * 16;  // key index
                ap.u[e] = Pt[w][l16][kk];
            }
#pragma unroll
            for (int j = 0; j < 4; ++j) {
                BFrag bv;
                int n = j * 16 + l16;                      // hd column
#pragma unroll
                for (int e = 0; e < 16; ++e) {
                    int kk = st * 32 + e + hh * 16;        // key index
                    bv.u[e] = VtT[buf][n][kk];             // contiguous -> ds_load_b128
                }
                o[j] = __builtin_amdgcn_wmma_f32_16x16x32_bf16(
                    false, ap.v, false, bv.v, (short)0, o[j], false, false);
            }
        }

        WAIT_ASYNC();          // next KV tile resident
        __syncthreads();       // all waves done with current buffer + Pt
        buf ^= 1;
    }

    // ---- normalize and store attention output (bf16, [S, D]) ----
#pragma unroll
    for (int j = 0; j < 4; ++j) {
#pragma unroll
        for (int v = 0; v < 8; ++v) {
            int srow = qw + hh * 8 + v;
            int col  = h * 64 + j * 16 + l16;
            Og[(size_t)srow * D + col] = f2bf(o[j][v] / rsum[v]);
        }
    }
}

// =====================================================================
// Host-side launch
// =====================================================================
extern "C" void kernel_launch(void* const* d_in, const int* in_sizes, int n_in,
                              void* d_out, int out_size, void* d_ws, size_t ws_size,
                              hipStream_t stream)
{
    const float* x  = (const float*)d_in[0];
    const float* Wq = (const float*)d_in[1];
    const float* bq = (const float*)d_in[2];
    const float* Wk = (const float*)d_in[3];
    const float* bk = (const float*)d_in[4];
    const float* Wv = (const float*)d_in[5];
    const float* bv = (const float*)d_in[6];
    const float* Wo = (const float*)d_in[7];
    const float* bo = (const float*)d_in[8];
    float* out = (float*)d_out;

    const int S = 2048, D = 1024, H = 16, HD = 64;
    const size_t SD = (size_t)S * D;   // 2M elems
    const size_t DD = (size_t)D * D;   // 1M elems

    unsigned short* qws = (unsigned short*)d_ws;       // [S,D] bf16
    unsigned short* kws = qws + SD;
    unsigned short* vws = kws + SD;
    unsigned short* ows = vws + SD;                    // attention output
    unsigned short* xbf = ows + SD;                    // x in bf16
    unsigned short* wqb = xbf + SD;                    // weights in bf16
    unsigned short* wkb = wqb + DD;
    unsigned short* wvb = wkb + DD;
    unsigned short* wob = wvb + DD;

    // one-shot fp32 -> bf16 conversions
    {
        int n4 = (int)(SD / 4);
        f32_to_bf16_kernel<<<(n4 + 255) / 256, 256, 0, stream>>>(x, xbf, n4);
        int w4 = (int)(DD / 4);
        f32_to_bf16_kernel<<<(w4 + 255) / 256, 256, 0, stream>>>(Wq, wqb, w4);
        f32_to_bf16_kernel<<<(w4 + 255) / 256, 256, 0, stream>>>(Wk, wkb, w4);
        f32_to_bf16_kernel<<<(w4 + 255) / 256, 256, 0, stream>>>(Wv, wvb, w4);
        f32_to_bf16_kernel<<<(w4 + 255) / 256, 256, 0, stream>>>(Wo, wob, w4);
    }

    dim3 gg(S / 128, D / 128);
    gemm_bt_bf16_kernel<false><<<gg, 256, 0, stream>>>(xbf, wqb, bq, nullptr, qws, S, D, D);
    gemm_bt_bf16_kernel<false><<<gg, 256, 0, stream>>>(xbf, wkb, bk, nullptr, kws, S, D, D);
    gemm_bt_bf16_kernel<false><<<gg, 256, 0, stream>>>(xbf, wvb, bv, nullptr, vws, S, D, D);

    int rope_total = 2 * S * H * (HD / 2);
    rope_kernel<<<(rope_total + 255) / 256, 256, 0, stream>>>(qws, kws, S, H, HD);

    dim3 ga(S / 64, H);
    flash_attn_kernel<<<ga, 128, 0, stream>>>(qws, kws, vws, ows, S, D);

    gemm_bt_bf16_kernel<true><<<gg, 256, 0, stream>>>(ows, wob, bo, out, nullptr, S, D, D);
}
